// SD_attn_parallel_16131897164216
// MI455X (gfx1250) — matrix operand order repaired and checked by reading, not compiled, not run on hardware
//
#include <hip/hip_runtime.h>
#include <hip/hip_bf16.h>

// ---------------------------------------------------------------------------
// MI455X (gfx1250, wave32) fused shifted-window attention.
//   K1: QKV GEMM (bf16 WMMA, double-buffered) + bias + 2D RoPE + scale
//   K2: per (window,head) attention with async global->LDS V staging
//   K3: proj GEMM (bf16 WMMA, double-buffered), window-unscatter+roll in gather
// Workspace requirement: ~302 MB (weights-T + q + k + v + y, all bf16).
// ---------------------------------------------------------------------------

typedef __attribute__((ext_vector_type(16))) __bf16        v16bf;
typedef __attribute__((ext_vector_type(8)))  float         v8f;
typedef __attribute__((ext_vector_type(8)))  unsigned int  v8u;

#define HEADS   8
#define HD      32
#define NTOK    64          // tokens per window (8x8)
#define DIM     256
#define QKVN    768
#define IMGH    192
#define IMGW    192
#define BS      4
#define TOKENS  (BS*IMGH*IMGW)   // 147456
#define NWIN    (TOKENS/NTOK)    // 2304
#define NWM     576
#define QSCALE  0.17677669529663687f   // 32^-0.5

__device__ __forceinline__ unsigned short f2bf(float f) {
    unsigned int u = __float_as_uint(f);
    u += 0x7FFFu + ((u >> 16) & 1u);          // round-to-nearest-even
    return (unsigned short)(u >> 16);
}

__device__ __forceinline__ v8f wmma_bf16(v16bf a, v16bf b, v8f c) {
    return __builtin_amdgcn_wmma_f32_16x16x32_bf16(false, a, false, b,
                                                   (short)0, c, false, false);
}

// A fragment (16x32 bf16): lane holds row M=lane&15; kbase=(lane>>4)*16;
// elements 0..15 at k=kbase+i (contiguous)
__device__ __forceinline__ v16bf load_a_frag(const unsigned short* rowbase, int kbase) {
    uint4 c0 = *(const uint4*)(rowbase + kbase);
    uint4 c1 = *(const uint4*)(rowbase + kbase + 8);
    v8u r; r[0]=c0.x; r[1]=c0.y; r[2]=c0.z; r[3]=c0.w;
           r[4]=c1.x; r[5]=c1.y; r[6]=c1.z; r[7]=c1.w;
    return __builtin_bit_cast(v16bf, r);
}

// B fragment (32x16 bf16): lane holds col N=lane&15; k=(lane>>4)*16 + i,
// i=0..15 contiguous -> 32 contiguous bytes at `p`.
__device__ __forceinline__ v16bf load_b_frag(const unsigned short* p) {
    uint4 c0 = *(const uint4*)(p);
    uint4 c1 = *(const uint4*)(p + 8);
    v8u r; r[0]=c0.x; r[1]=c0.y; r[2]=c0.z; r[3]=c0.w;
           r[4]=c1.x; r[5]=c1.y; r[6]=c1.z; r[7]=c1.w;
    return __builtin_bit_cast(v16bf, r);
}

// raw fp32 A-fragment load (prefetchable) + deferred bf16 conversion
__device__ __forceinline__ void load_a_raw(const float* xrow, int ks, int kbase,
                                           float4& f0, float4& f1,
                                           float4& f2, float4& f3) {
    f0 = *(const float4*)(xrow + ks*32 + kbase);
    f1 = *(const float4*)(xrow + ks*32 + kbase + 4);
    f2 = *(const float4*)(xrow + ks*32 + kbase + 8);
    f3 = *(const float4*)(xrow + ks*32 + kbase + 12);
}
__device__ __forceinline__ v16bf cvt_a(float4 f0, float4 f1, float4 f2, float4 f3) {
    v8u au;
    au[0] = (unsigned)f2bf(f0.x) | ((unsigned)f2bf(f0.y) << 16);
    au[1] = (unsigned)f2bf(f0.z) | ((unsigned)f2bf(f0.w) << 16);
    au[2] = (unsigned)f2bf(f1.x) | ((unsigned)f2bf(f1.y) << 16);
    au[3] = (unsigned)f2bf(f1.z) | ((unsigned)f2bf(f1.w) << 16);
    au[4] = (unsigned)f2bf(f2.x) | ((unsigned)f2bf(f2.y) << 16);
    au[5] = (unsigned)f2bf(f2.z) | ((unsigned)f2bf(f2.w) << 16);
    au[6] = (unsigned)f2bf(f3.x) | ((unsigned)f2bf(f3.y) << 16);
    au[7] = (unsigned)f2bf(f3.z) | ((unsigned)f2bf(f3.w) << 16);
    return __builtin_bit_cast(v16bf, au);
}

// Shift-mask height region (reference mask collapses to h-region only:
// 0 for h<184, 1 for 184..187, 2 for >=188; only tile row nh==11 is affected).
__device__ __forceinline__ int hregion(int nh, int dh, int i) {
    if (nh != 11) return 0;
    int h = 176 + 2*i + dh;
    return (h < 184) ? 0 : ((h < 188) ? 1 : 2);
}

// ---------------------------------------------------------------------------
// K0: transpose weights to bf16: WqkvT[n][k], WprojT[n][k]
// ---------------------------------------------------------------------------
__global__ void prep_weights_kernel(const float* __restrict__ Wqkv,
                                    const float* __restrict__ Wproj,
                                    unsigned short* __restrict__ WqkvT,
                                    unsigned short* __restrict__ WprojT) {
    int i = blockIdx.x * blockDim.x + threadIdx.x;
    if (i < DIM * QKVN) {
        int n = i / DIM, kk = i % DIM;
        WqkvT[i] = f2bf(Wqkv[kk * QKVN + n]);
    } else {
        int j = i - DIM * QKVN;
        if (j < DIM * DIM) {
            int n = j / DIM, kk = j % DIM;
            WprojT[j] = f2bf(Wproj[kk * DIM + n]);
        }
    }
}

// ---------------------------------------------------------------------------
// K1: qkv = x @ Wqkv + b, fused RoPE(q,k) + q*SCALE; outputs bf16
//     q/k/v layout: [win][head][tok][d]  (d contiguous)
// block = 256 thr (8 waves); wave tile = 16 rows x 64 cols; block = 128 rows
// grid = (TOKENS/128, 768/64). Inner loop double-buffered so WMMAs only wait
// on already-aged loads (next k-step's 12 loads stay in flight).
// ---------------------------------------------------------------------------
__global__ __launch_bounds__(256) void qkv_rope_kernel(
    const float* __restrict__ x, const unsigned short* __restrict__ WqkvT,
    const float* __restrict__ b_qkv,
    unsigned short* __restrict__ qo, unsigned short* __restrict__ ko,
    unsigned short* __restrict__ vo)
{
    __shared__ unsigned short stage[8][16][64];   // 16 KB

    const int lane = threadIdx.x & 31;
    const int wave = threadIdx.x >> 5;
    const int rowW = blockIdx.x * 128 + wave * 16;   // token base of wave strip
    const int n0   = blockIdx.y * 64;                // output col base
    const int half = lane >> 4;
    const int kbase = half * 16;

    const float* xrow = x + (size_t)(rowW + (lane & 15)) * DIM;
    const unsigned short* bbase =
        WqkvT + (size_t)(n0 + (lane & 15)) * DIM + half * 16;

    v8f acc[4];
    #pragma unroll
    for (int t = 0; t < 4; ++t) acc[t] = v8f{0,0,0,0,0,0,0,0};

    // prologue: fragments for ks = 0
    float4 a0, a1, a2, a3;
    load_a_raw(xrow, 0, kbase, a0, a1, a2, a3);
    v16bf bcur[4];
    #pragma unroll
    for (int t = 0; t < 4; ++t) bcur[t] = load_b_frag(bbase + t * (16 * DIM));

    #pragma unroll
    for (int ks = 0; ks < 7; ++ks) {
        // prefetch ks+1
        float4 n_a0, n_a1, n_a2, n_a3;
        load_a_raw(xrow, ks + 1, kbase, n_a0, n_a1, n_a2, n_a3);
        v16bf bnxt[4];
        #pragma unroll
        for (int t = 0; t < 4; ++t)
            bnxt[t] = load_b_frag(bbase + t * (16 * DIM) + (ks + 1) * 32);
        // compute ks
        v16bf a = cvt_a(a0, a1, a2, a3);
        #pragma unroll
        for (int t = 0; t < 4; ++t) acc[t] = wmma_bf16(a, bcur[t], acc[t]);
        a0 = n_a0; a1 = n_a1; a2 = n_a2; a3 = n_a3;
        #pragma unroll
        for (int t = 0; t < 4; ++t) bcur[t] = bnxt[t];
    }
    {   // epilogue: ks = 7
        v16bf a = cvt_a(a0, a1, a2, a3);
        #pragma unroll
        for (int t = 0; t < 4; ++t) acc[t] = wmma_bf16(a, bcur[t], acc[t]);
    }

    const int sec = n0 >> 8;                 // 0=q 1=k 2=v
    const int h0  = (n0 & 255) >> 5;         // first head of this 64-col span
    unsigned short* dst = (sec == 0) ? qo : ((sec == 1) ? ko : vo);

    #pragma unroll
    for (int t = 0; t < 4; ++t) {
        int ncol = n0 + t*16 + (lane & 15);
        float bias = b_qkv[ncol];
        int d = ncol & 31;
        int j = d & 7;
        float invf = __powf(10000.0f, -(float)j * 0.125f);
        bool useH = (d & 16) == 0;
        bool hi   = (d & 8) != 0;
        #pragma unroll
        for (int r = 0; r < 8; ++r) {
            float val = acc[t][r] + bias;
            if (sec < 2) {     // RoPE on q,k; pair partner sits in lane^8
                int tok = rowW + r + 8*half;
                int nwi = tok & 63;
                float pos = useH ? (float)(nwi >> 3) : (float)(nwi & 7);
                float sn, cs;
                __sincosf(pos * invf, &sn, &cs);
                float other = __shfl_xor(val, 8, 32);
                float o = hi ? (other * sn + val * cs) : (val * cs - other * sn);
                if (sec == 0) o *= QSCALE;
                val = o;
            }
            stage[wave][r + 8*half][t*16 + (lane & 15)] = f2bf(val);
        }
    }
    __syncthreads();

    // coalesced write-out: lane -> (token = lane>>1, head-half = lane&1)
    {
        int tok16 = lane >> 1;
        int hh    = lane & 1;
        int tokg  = rowW + tok16;
        int bwin  = tokg >> 6;
        int nwi   = tokg & 63;
        size_t base = (((size_t)bwin * HEADS + h0 + hh) * NTOK + nwi) * HD;
        const uint4* src = (const uint4*)&stage[wave][tok16][hh * 32];
        uint4* dp = (uint4*)(dst + base);
        dp[0] = src[0]; dp[1] = src[1]; dp[2] = src[2]; dp[3] = src[3];
    }
}

// ---------------------------------------------------------------------------
// K2: attention for one (window, head). block = 128 thr (4 waves);
//     wave w handles query strip [16w,16w+16). grid = (NWIN, HEADS)
//     V tile staged with GLOBAL_LOAD_ASYNC_TO_LDS_B128 (ASYNCcnt-tracked,
//     bypasses VGPRs entirely — CDNA5 async memory path, ISA §15.18.3 op 98).
// ---------------------------------------------------------------------------
__global__ __launch_bounds__(128) void attn_kernel(
    const unsigned short* __restrict__ q, const unsigned short* __restrict__ k,
    const unsigned short* __restrict__ v, unsigned short* __restrict__ y)
{
    __shared__ unsigned short vt[64][32];   // V tile [tok][d]   4 KB
    __shared__ unsigned short pt[64][64];   // P (softmax out)   8 KB
    __shared__ unsigned short ot[64][32];   // O staging         4 KB

    const int bwin = blockIdx.x;
    const int head = blockIdx.y;
    const int lane = threadIdx.x & 31;
    const int wave = threadIdx.x >> 5;
    const int half = lane >> 4;
    const int kbase = half * 16;
    const size_t base = ((size_t)bwin * HEADS + head) * NTOK * HD;

    // ---- async V stage: memory -> LDS, 16B per lane, 2 rounds (4 KB) ------
    {
        unsigned lds0 = (unsigned)(size_t)(&vt[0][0]) + (unsigned)threadIdx.x * 16u;
        const unsigned short* g0 = v + base + (size_t)threadIdx.x * 8;
        asm volatile("global_load_async_to_lds_b128 %0, %1, off"
                     :: "v"(lds0), "v"(g0) : "memory");
        unsigned lds1 = lds0 + 2048u;
        const unsigned short* g1 = g0 + 1024;
        asm volatile("global_load_async_to_lds_b128 %0, %1, off"
                     :: "v"(lds1), "v"(g1) : "memory");
        asm volatile("s_wait_asynccnt 0" ::: "memory");
    }
    __syncthreads();

    // ---- S = (q*scale) @ k^T : 4 WMMA, K=32 in one step -------------------
    const int mtok = wave * 16 + (lane & 15);
    v16bf qa = load_a_frag(q + base + (size_t)mtok * HD, kbase);

    v8f s[4];
    #pragma unroll
    for (int jt = 0; jt < 4; ++jt) {
        int key = jt*16 + (lane & 15);
        v16bf kb = load_b_frag(k + base + (size_t)key * HD + half * 16);
        v8f z = v8f{0,0,0,0,0,0,0,0};
        s[jt] = wmma_bf16(qa, kb, z);
    }

    // ---- analytic shift mask ---------------------------------------------
    const int mw   = bwin % NWM;
    const int tile = mw >> 2;
    const int nh   = tile / 12;
    const int dh   = (mw >> 1) & 1;
    const int iq   = 2*wave + half;              // query window-row
    const int rq   = hregion(nh, dh, iq);
    #pragma unroll
    for (int jt = 0; jt < 4; ++jt) {
        int ik = 2*jt + ((lane >> 3) & 1);       // key window-row
        float mterm = (hregion(nh, dh, ik) != rq) ? -1e30f : 0.0f;
        #pragma unroll
        for (int r = 0; r < 8; ++r) s[jt][r] += mterm;
    }

    // ---- softmax (row = 16 lanes of a half-group x 4 tiles) ---------------
    #pragma unroll
    for (int r = 0; r < 8; ++r) {
        float mx = fmaxf(fmaxf(s[0][r], s[1][r]), fmaxf(s[2][r], s[3][r]));
        mx = fmaxf(mx, __shfl_xor(mx, 1, 32));
        mx = fmaxf(mx, __shfl_xor(mx, 2, 32));
        mx = fmaxf(mx, __shfl_xor(mx, 4, 32));
        mx = fmaxf(mx, __shfl_xor(mx, 8, 32));
        float sm = 0.0f;
        #pragma unroll
        for (int jt = 0; jt < 4; ++jt) {
            float e = __expf(s[jt][r] - mx);
            s[jt][r] = e; sm += e;
        }
        sm += __shfl_xor(sm, 1, 32);
        sm += __shfl_xor(sm, 2, 32);
        sm += __shfl_xor(sm, 4, 32);
        sm += __shfl_xor(sm, 8, 32);
        float rinv = 1.0f / sm;
        int row = wave*16 + r + 8*half;
        #pragma unroll
        for (int jt = 0; jt < 4; ++jt)
            pt[row][jt*16 + (lane & 15)] = f2bf(s[jt][r] * rinv);
    }
    __syncthreads();

    // ---- O = P @ V : K=64 -> 2 k-steps, N=32 -> 2 n-tiles (4 WMMA) --------
    v8f o[2] = { v8f{0,0,0,0,0,0,0,0}, v8f{0,0,0,0,0,0,0,0} };
    #pragma unroll
    for (int ks = 0; ks < 2; ++ks) {
        v16bf pa = load_a_frag(&pt[wave*16 + (lane & 15)][ks*32], kbase);
        #pragma unroll
        for (int nt = 0; nt < 2; ++nt) {
            int d  = nt*16 + (lane & 15);
            int k0 = ks*32 + half*16;
            v8u bu;
            #pragma unroll
            for (int w = 0; w < 8; ++w) {
                unsigned lo = vt[k0 + 2*w][d];
                unsigned hb = vt[k0 + 2*w + 1][d];
                bu[w] = lo | (hb << 16);
            }
            o[nt] = wmma_bf16(pa, __builtin_bit_cast(v16bf, bu), o[nt]);
        }
    }

    #pragma unroll
    for (int nt = 0; nt < 2; ++nt)
        #pragma unroll
        for (int r = 0; r < 8; ++r)
            ot[wave*16 + r + 8*half][nt*16 + (lane & 15)] = f2bf(o[nt][r]);
    __syncthreads();

    // coalesced write: y[win][tok][head*32 + d]
    {
        int tok = threadIdx.x >> 1;
        int hh  = threadIdx.x & 1;
        const uint4* src = (const uint4*)&ot[tok][hh * 16];
        uint4* dp = (uint4*)(y + ((size_t)bwin * NTOK + tok) * DIM + head * HD + hh * 16);
        dp[0] = src[0]; dp[1] = src[1];
    }
}

// ---------------------------------------------------------------------------
// K3: out = gather(y) @ Wproj + b_proj  (fp32 output)
//     Gather folds window-unscatter + roll(4,4): rows stay contiguous 512B.
// block = 256 thr (8 waves); wave = 16 rows x 64 cols; grid = (TOKENS/128, 4)
// Double-buffered like K1.
// ---------------------------------------------------------------------------
__global__ __launch_bounds__(256) void proj_kernel(
    const unsigned short* __restrict__ y, const unsigned short* __restrict__ WprojT,
    const float* __restrict__ b_proj, float* __restrict__ out)
{
    const int lane = threadIdx.x & 31;
    const int wave = threadIdx.x >> 5;
    const int rowW = blockIdx.x * 128 + wave * 16;
    const int n0   = blockIdx.y * 64;
    const int half = lane >> 4;
    const int kbase = half * 16;

    // inverse roll + window scatter for this lane's A row
    int timg = rowW + (lane & 15);
    int bs   = timg / (IMGH * IMGW);
    int rem  = timg - bs * (IMGH * IMGW);
    int hI   = rem / IMGW;
    int wI   = rem - hI * IMGW;
    int hp = hI - 4; if (hp < 0) hp += IMGH;
    int wp = wI - 4; if (wp < 0) wp += IMGW;
    int nh = hp >> 4, th = hp & 15, ii = th >> 1, d_h = th & 1;
    int nw = wp >> 4, tw = wp & 15, jj = tw >> 1, d_w = tw & 1;
    int bwin = ((bs * 144 + nh * 12 + nw) << 2) + d_h * 2 + d_w;
    int nwi  = ii * 8 + jj;
    const unsigned short* yrow = y + ((size_t)bwin * NTOK + nwi) * DIM;
    const unsigned short* bbase =
        WprojT + (size_t)(n0 + (lane & 15)) * DIM + half * 16;

    v8f acc[4];
    #pragma unroll
    for (int t = 0; t < 4; ++t) acc[t] = v8f{0,0,0,0,0,0,0,0};

    v16bf acur = load_a_frag(yrow, kbase);
    v16bf bcur[4];
    #pragma unroll
    for (int t = 0; t < 4; ++t) bcur[t] = load_b_frag(bbase + t * (16 * DIM));

    #pragma unroll
    for (int ks = 0; ks < 7; ++ks) {
        v16bf anxt = load_a_frag(yrow + (ks + 1) * 32, kbase);
        v16bf bnxt[4];
        #pragma unroll
        for (int t = 0; t < 4; ++t)
            bnxt[t] = load_b_frag(bbase + t * (16 * DIM) + (ks + 1) * 32);
        #pragma unroll
        for (int t = 0; t < 4; ++t) acc[t] = wmma_bf16(acur, bcur[t], acc[t]);
        acur = anxt;
        #pragma unroll
        for (int t = 0; t < 4; ++t) bcur[t] = bnxt[t];
    }
    #pragma unroll
    for (int t = 0; t < 4; ++t) acc[t] = wmma_bf16(acur, bcur[t], acc[t]);

    #pragma unroll
    for (int t = 0; t < 4; ++t) {
        int n = n0 + t*16 + (lane & 15);
        float bias = b_proj[n];
        #pragma unroll
        for (int r = 0; r < 8; ++r)
            out[(size_t)(rowW + r + 8*half) * DIM + n] = acc[t][r] + bias;
    }
}

// ---------------------------------------------------------------------------
extern "C" void kernel_launch(void* const* d_in, const int* in_sizes, int n_in,
                              void* d_out, int out_size, void* d_ws, size_t ws_size,
                              hipStream_t stream) {
    const float* x      = (const float*)d_in[0];
    const float* W_qkv  = (const float*)d_in[1];
    const float* b_qkv  = (const float*)d_in[2];
    const float* W_proj = (const float*)d_in[3];
    const float* b_proj = (const float*)d_in[4];
    float* out = (float*)d_out;

    char* ws = (char*)d_ws;
    const size_t SZ_WQKVT = (size_t)QKVN * DIM * 2;        //   384 KB
    const size_t SZ_WPROJ = (size_t)DIM * DIM * 2;         //   128 KB
    const size_t SZ_QKV   = (size_t)NWIN * HEADS * NTOK * HD * 2;  // 72 MB each
    unsigned short* WqkvT  = (unsigned short*)(ws);
    unsigned short* WprojT = (unsigned short*)(ws + SZ_WQKVT);
    unsigned short* qb     = (unsigned short*)(ws + SZ_WQKVT + SZ_WPROJ);
    unsigned short* kb     = (unsigned short*)(ws + SZ_WQKVT + SZ_WPROJ + SZ_QKV);
    unsigned short* vb     = (unsigned short*)(ws + SZ_WQKVT + SZ_WPROJ + 2*SZ_QKV);
    unsigned short* yb     = (unsigned short*)(ws + SZ_WQKVT + SZ_WPROJ + 3*SZ_QKV);

    prep_weights_kernel<<<(DIM*QKVN + DIM*DIM + 255) / 256, 256, 0, stream>>>(
        W_qkv, W_proj, WqkvT, WprojT);

    qkv_rope_kernel<<<dim3(TOKENS / 128, QKVN / 64), 256, 0, stream>>>(
        x, WqkvT, b_qkv, qb, kb, vb);

    attn_kernel<<<dim3(NWIN, HEADS), 128, 0, stream>>>(qb, kb, vb, yb);

    proj_kernel<<<dim3(TOKENS / 128, DIM / 64), 256, 0, stream>>>(
        yb, WprojT, b_proj, out);
}
